// CalculateFlow_62801011802519
// MI455X (gfx1250) — compile-verified, matrix operand order checked by reading, not run on hardware
//
#include <hip/hip_runtime.h>

// ---------------------------------------------------------------------------
// CalculateFlow on MI455X (gfx1250, wave32)
//
// Pipeline (matches reference):
//   1) prep:    binomial 3x3 (replicate pad) -> fb (f32), quantize to 0..255
//               stored into ZERO-padded f16 buffers fqp (pad 5) / gqp (pad 2)
//   2) flow:    per 16x8 tile: TDM tensor_load_to_lds stages f/g tiles in LDS;
//               per wave (16-pixel strip) 49 SADs whose 25-tap reductions run
//               on the matrix pipe via v_wmma_f32_16x16x32_f16 (B = ones,
//               exact: integer f16 inputs <=255, f32 accumulate); lexicographic
//               (cost*64 + spiral rank) argmin; LK subpixel refine -> flow_raw
//   3) median:  3x3 median (replicate pad), sorting network
//   4) bilateral: 5x5 guided filter (zero pad), writes d_out
// ---------------------------------------------------------------------------

#define HH 256
#define WW 256
#define PF 5            // fq pad = SR + K/2 = 3 + 2
#define PG 2            // gq pad = K/2
#define FQ_W 272        // padded row stride (zero-filled to stride)
#define FQ_H 266        // 256 + 2*PF
#define GQ_H 260        // 256 + 2*PG

#define LF_W 28         // f LDS tile: 18 rows x 28 cols (need 18x26)
#define LF_H 18
#define LG_W 24         // g LDS tile: 12 rows x 24 cols (need 12x20)
#define LG_H 12

typedef _Float16 v16h __attribute__((ext_vector_type(16)));
typedef float    v8f  __attribute__((ext_vector_type(8)));
typedef unsigned int v4u __attribute__((ext_vector_type(4)));
typedef int      v4i  __attribute__((ext_vector_type(4)));
typedef int      v8i  __attribute__((ext_vector_type(8)));

// spiral rank of displacement (dy,dx), flat index (dy+3)*7+(dx+3)
__constant__ int SPIRAL49[49] = {
  42,43,44,45,46,47,48,
  41,20,21,22,23,24,25,
  40,19, 6, 7, 8, 9,26,
  39,18, 5, 0, 1,10,27,
  38,17, 4, 3, 2,11,28,
  37,16,15,14,13,12,29,
  36,35,34,33,32,31,30 };

__device__ __forceinline__ float binom3(const float* __restrict__ img, int x, int y) {
  int ym = y > 0 ? y - 1 : 0, yp = y < HH - 1 ? y + 1 : HH - 1;
  int xm = x > 0 ? x - 1 : 0, xp = x < WW - 1 ? x + 1 : WW - 1;
  float r0 = img[ym*WW+xm] + 2.f*img[ym*WW+x] + img[ym*WW+xp];
  float r1 = img[y *WW+xm] + 2.f*img[y *WW+x] + img[y *WW+xp];
  float r2 = img[yp*WW+xm] + 2.f*img[yp*WW+x] + img[yp*WW+xp];
  return (r0 + 2.f*r1 + r2) * (1.f/16.f);
}

// ---------------- kernel 1: binomial + quantize into padded f16 -------------
__global__ void prep_kernel(const float* __restrict__ f, const float* __restrict__ g,
                            float* __restrict__ fb, _Float16* __restrict__ fqp,
                            _Float16* __restrict__ gqp) {
  int px = blockIdx.x * 16 + threadIdx.x;
  int py = blockIdx.y * 16 + threadIdx.y;
  if (px >= FQ_W || py >= FQ_H) return;
  // f side (pad 5)
  {
    int x = px - PF, y = py - PF;
    float q = 0.f;
    if (x >= 0 && x < WW && y >= 0 && y < HH) {
      float v = binom3(f, x, y);
      fb[y*WW + x] = v;
      q = fminf(fmaxf(rintf(v * 255.f), 0.f), 255.f);   // rintf = round-half-even
    }
    fqp[py*FQ_W + px] = (_Float16)q;
  }
  // g side (pad 2)
  if (py < GQ_H) {
    int x = px - PG, y = py - PG;
    float q = 0.f;
    if (x >= 0 && x < WW && y >= 0 && y < HH) {
      float v = binom3(g, x, y);
      q = fminf(fmaxf(rintf(v * 255.f), 0.f), 255.f);
    }
    gqp[py*FQ_W + px] = (_Float16)q;
  }
}

// ---------------- kernel 2: TDM-staged SAD + WMMA reduce + argmin + subpixel
__global__ __launch_bounds__(256)
void flow_kernel(const _Float16* __restrict__ fqp, const _Float16* __restrict__ gqp,
                 const float* __restrict__ fb, float* __restrict__ flow_raw) {
  __shared__ _Float16 lds_f[LF_H * LF_W];
  __shared__ _Float16 lds_g[LG_H * LG_W];
  __shared__ int lds_best[8 * 16];

  const int tid  = threadIdx.x;
  const int wid  = tid >> 5;          // wave id 0..7 (wave32)
  const int lane = tid & 31;
  const int bx = blockIdx.x, by = blockIdx.y;

  // ---- Tensor Data Mover: wave 0 DMAs both tiles into LDS -----------------
  if (wid == 0) {
    unsigned lf = (unsigned)(unsigned long long)(void*)lds_f;
    unsigned lg = (unsigned)(unsigned long long)(void*)lds_g;
    unsigned long long gaf =
        (unsigned long long)(const void*)(fqp + (size_t)(by * 8) * FQ_W + bx * 16);
    unsigned long long gag =
        (unsigned long long)(const void*)(gqp + (size_t)(by * 8) * FQ_W + bx * 16);
    // D# group0: count=1 | lds_addr | global_addr[31:0] | global[56:32], type=2
    v4u g0f = { 1u, lf, (unsigned)gaf,
                (unsigned)((gaf >> 32) & 0x01FFFFFFu) | (2u << 30) };
    v4u g0g = { 1u, lg, (unsigned)gag,
                (unsigned)((gag >> 32) & 0x01FFFFFFu) | (2u << 30) };
    // D# group1: data_size=1(2B) | tdim0 | tdim1 | tile0 | tile1 | stride0
    v8i g1f = { (int)(1u << 16), (int)(LF_W << 16), (int)(LF_H << 16),
                (int)(LF_W << 16), LF_H, FQ_W, 0, 0 };
    v8i g1g = { (int)(1u << 16), (int)(LG_W << 16), (int)(LG_H << 16),
                (int)(LG_W << 16), LG_H, FQ_W, 0, 0 };
    v4i z4 = {0, 0, 0, 0};
#if defined(__clang_major__) && (__clang_major__ >= 23)
    v8i z8 = {0, 0, 0, 0, 0, 0, 0, 0};
    __builtin_amdgcn_tensor_load_to_lds(g0f, g1f, z4, z4, z8, 0);
    __builtin_amdgcn_tensor_load_to_lds(g0g, g1g, z4, z4, z8, 0);
#else
    __builtin_amdgcn_tensor_load_to_lds(g0f, g1f, z4, z4, 0);
    __builtin_amdgcn_tensor_load_to_lds(g0g, g1g, z4, z4, 0);
#endif
    __builtin_amdgcn_s_wait_tensorcnt(0);
  }
  __syncthreads();

  // ---- per-wave 16-pixel strip: SAD via WMMA ------------------------------
  // A (16x32 f16) layout: lanes 0-15 row M=lane hold K 0-7,16-23;
  //                       lanes 16-31 row M=lane-16 hold K 8-15,24-31.
  const int j  = lane & 15;     // A row == pixel within strip
  const int hi = lane >> 4;
  const int w  = wid;           // strip row within tile

  int   Koff[16];
  float gv[16];
#pragma unroll
  for (int h = 0; h < 16; ++h) {
    int K = (h < 8 ? h : h + 8) + (hi ? 8 : 0);
    if (K < 25) {
      int ky = K / 5, kx = K % 5;
      gv[h]   = (float)lds_g[(w + ky) * LG_W + (j + kx)];
      Koff[h] = (w + ky + 3) * LF_W + (j + kx + 3);
    } else { gv[h] = 0.f; Koff[h] = -1; }
  }
  v16h bones;
#pragma unroll
  for (int h = 0; h < 16; ++h) bones[h] = (_Float16)1.0f;
  v8f cz = {};

  int best[8], bidx[8];
#pragma unroll
  for (int t = 0; t < 8; ++t) { best[t] = 0x7fffffff; bidx[t] = 0; }

  for (int d = 0; d < 49; ++d) {
    int dy = d / 7 - 3, dx = d % 7 - 3;
    int dof = dy * LF_W + dx;
    v16h aa;
#pragma unroll
    for (int h = 0; h < 16; ++h) {
      float fv = (Koff[h] >= 0) ? (float)lds_f[Koff[h] + dof] : 0.f;
      aa[h] = (_Float16)fabsf(fv - gv[h]);          // exact: integers <= 255
    }
    v8f c = __builtin_amdgcn_wmma_f32_16x16x32_f16(
        false, aa, false, bones, (short)0, cz, false, false);
    // lane holds costs of pixels hi*8 .. hi*8+7 in c[0..7] (columns identical)
    int rk = SPIRAL49[d];
#pragma unroll
    for (int t = 0; t < 8; ++t) {
      int sc = ((int)c[t]) * 64 + rk;               // lexicographic score
      if (sc < best[t]) { best[t] = sc; bidx[t] = d; }
    }
  }

  // redistribute best displacement: one representative lane per column group
  if (lane == 0 || lane == 16) {
    int base = wid * 16 + hi * 8;
#pragma unroll
    for (int t = 0; t < 8; ++t) lds_best[base + t] = bidx[t];
  }
  __syncthreads();
  int mb = lds_best[wid * 16 + j];

  // ---- subpixel (LK on border ring of 5x5 patch) + store ------------------
  if (lane < 16) {
    int Y = by * 8 + wid, X = bx * 16 + lane;
    int dy = mb / 7 - 3, dx = mb % 7 - 3;
    float a = 0.f, b = 0.f, dd = 0.f, p = 0.f, q = 0.f;
#pragma unroll
    for (int t = 0; t < 25; ++t) {
      int ky = t / 5, kx = t % 5;
      if (ky != 0 && ky != 4 && kx != 0 && kx != 4) continue;   // ring mask
      float fT = (float)lds_f[(w + dy + ky + 3) * LF_W + (lane + dx + kx + 3)]
                 * (1.f / 255.f);
      float gT = (float)lds_g[(w + ky) * LG_W + (lane + kx)] * (1.f / 255.f);
      int yy = Y + ky - 2, xx = X + kx - 2;
      float dfy = 0.f, dfx = 0.f;
      if (yy >= 0 && yy < HH && xx >= 0 && xx < WW) {           // zero-pad tmpl
        int ym = yy > 0 ? yy - 1 : 0, yp = yy < HH - 1 ? yy + 1 : HH - 1;
        int xm = xx > 0 ? xx - 1 : 0, xp = xx < WW - 1 ? xx + 1 : WW - 1;
        dfy = (fb[yp * WW + xx] - fb[ym * WW + xx]) * 0.5f;
        dfx = (fb[yy * WW + xp] - fb[yy * WW + xm]) * 0.5f;
      }
      float z = gT - fT;
      a += dfx * dfx; b += dfx * dfy; dd += dfy * dfy;
      p += z * dfx;   q += z * dfy;
    }
    float det = a * dd - b * b;
    float su = 0.f, sv = 0.f;
    if (det > 1e-7f) {
      float inv = 1.f / det;
      float u = (dd * p - b * q) * inv;
      float v = (a * q - b * p) * inv;
      su = (fabsf(u) >= 1.f) ? 0.f : u;
      sv = (fabsf(v) >= 1.f) ? 0.f : v;
    }
    int idx = Y * WW + X;
    flow_raw[idx]            = (float)(-dy) + sv;   // channel 0 = y
    flow_raw[HH * WW + idx]  = (float)(-dx) + su;   // channel 1 = x
  }
}

// ---------------- kernel 3: 3x3 median (replicate pad) ----------------------
__device__ __forceinline__ void mmsort(float& a, float& b) {
  float t = fminf(a, b); b = fmaxf(a, b); a = t;
}
__global__ void median_kernel(const float* __restrict__ fin, float* __restrict__ fout) {
  int i = blockIdx.x * blockDim.x + threadIdx.x;
  if (i >= 2 * HH * WW) return;
  int c = i >> 16;             // HH*WW == 65536
  int pix = i & 65535;
  int y = pix >> 8, x = pix & 255;
  const float* src = fin + c * HH * WW;
  float v[9]; int n = 0;
#pragma unroll
  for (int ddy = -1; ddy <= 1; ++ddy)
#pragma unroll
    for (int ddx = -1; ddx <= 1; ++ddx) {
      int yy = min(max(y + ddy, 0), HH - 1);
      int xx = min(max(x + ddx, 0), WW - 1);
      v[n++] = src[yy * WW + xx];
    }
  mmsort(v[1],v[2]); mmsort(v[4],v[5]); mmsort(v[7],v[8]);
  mmsort(v[0],v[1]); mmsort(v[3],v[4]); mmsort(v[6],v[7]);
  mmsort(v[1],v[2]); mmsort(v[4],v[5]); mmsort(v[7],v[8]);
  mmsort(v[0],v[3]); mmsort(v[5],v[8]); mmsort(v[4],v[7]);
  mmsort(v[3],v[6]); mmsort(v[1],v[4]); mmsort(v[2],v[5]);
  mmsort(v[4],v[7]); mmsort(v[4],v[2]); mmsort(v[6],v[4]);
  mmsort(v[4],v[2]);
  fout[i] = v[4];
}

// ---------------- kernel 4: 5x5 bilateral (zero pad), writes d_out ----------
__global__ void bilateral_kernel(const float* __restrict__ fb,
                                 const float* __restrict__ vec,
                                 float* __restrict__ out) {
  int pix = blockIdx.x * blockDim.x + threadIdx.x;
  if (pix >= HH * WW) return;
  int y = pix >> 8, x = pix & 255;
  float cp = fb[pix];
  const float KC  = -1.f / 4.5f;   // -1/(2*1.5^2)
  const float SII = 200.f;         //  1/(2*0.05^2)
  float n0 = 0.f, n1 = 0.f, den = 0.f;
#pragma unroll
  for (int ddy = -2; ddy <= 2; ++ddy)
#pragma unroll
    for (int ddx = -2; ddx <= 2; ++ddx) {
      int yy = y + ddy, xx = x + ddx;
      bool in = (yy >= 0 && yy < HH && xx >= 0 && xx < WW);
      float ck = in ? fb[yy * WW + xx] : 0.f;
      float v0 = in ? vec[yy * WW + xx] : 0.f;
      float v1 = in ? vec[HH * WW + yy * WW + xx] : 0.f;
      float d2 = (cp - ck) * (cp - ck);
      float co = 1.f - fabsf(KC - d2 * SII);
      co = fminf(fmaxf(co, 0.f), 1.f);
      n0 += v0 * co; n1 += v1 * co; den += co;
    }
  out[pix]           = n0 / den;
  out[HH * WW + pix] = n1 / den;
}

// ---------------------------------------------------------------------------
extern "C" void kernel_launch(void* const* d_in, const int* in_sizes, int n_in,
                              void* d_out, int out_size, void* d_ws, size_t ws_size,
                              hipStream_t stream) {
  (void)in_sizes; (void)n_in; (void)out_size; (void)ws_size;
  const float* f_img = (const float*)d_in[0];
  const float* g_img = (const float*)d_in[1];

  char* ws = (char*)d_ws;                       // ~1.6 MB total
  float*    fb       = (float*)ws;                               // 262144 B
  _Float16* fqp      = (_Float16*)(ws + 262144);                 // 144704 B
  _Float16* gqp      = (_Float16*)(ws + 262144 + 144896);        // 141440 B
  float*    flow_raw = (float*)(ws + 262144 + 144896 + 141568);  // 524288 B
  float*    flow_med = flow_raw + 2 * HH * WW;                   // 524288 B

  dim3 b1(16, 16), g1((FQ_W + 15) / 16, (FQ_H + 15) / 16);
  prep_kernel<<<g1, b1, 0, stream>>>(f_img, g_img, fb, fqp, gqp);

  dim3 g2(WW / 16, HH / 8);
  flow_kernel<<<g2, 256, 0, stream>>>(fqp, gqp, fb, flow_raw);

  median_kernel<<<(2 * HH * WW) / 256, 256, 0, stream>>>(flow_raw, flow_med);

  bilateral_kernel<<<(HH * WW) / 256, 256, 0, stream>>>(fb, flow_med,
                                                        (float*)d_out);
}
